// MPAD_82532091560282
// MI455X (gfx1250) — matrix operand order repaired and checked by reading, not compiled, not run on hardware
//
#include <hip/hip_runtime.h>

#define BF16 __bf16

typedef __attribute__((ext_vector_type(16))) BF16 bf16x16;
typedef __attribute__((ext_vector_type(8)))  BF16 bf16x8;
typedef __attribute__((ext_vector_type(8)))  float f32x8;
typedef __attribute__((ext_vector_type(4)))  unsigned int u32x4;
typedef __attribute__((ext_vector_type(8)))  int i32x8;
typedef __attribute__((ext_vector_type(4)))  int i32x4;

union ABFrag { bf16x16 v; bf16x8 h8[2]; };

// ---------------- convert / gather / pad ----------------
__global__ void k_convert_pad(const float* __restrict__ src, BF16* __restrict__ dst,
                              int srows, int scols, int drows, int dcols)
{
    long i = (long)blockIdx.x * blockDim.x + threadIdx.x;
    long total = (long)drows * dcols;
    if (i >= total) return;
    int r = (int)(i / dcols), c = (int)(i % dcols);
    float v = (r < srows && c < scols) ? src[(long)r * scols + c] : 0.0f;
    dst[i] = (BF16)v;
}

__global__ void k_gather_cast(const int* __restrict__ xi, const float* __restrict__ emb,
                              BF16* __restrict__ h, long T, int F, int Fp)
{
    long i = (long)blockIdx.x * blockDim.x + threadIdx.x;
    long total = T * (long)Fp;
    if (i >= total) return;
    long r = i / Fp; int c = (int)(i % Fp);
    float v = (c < F) ? emb[(long)xi[r] * F + c] : 0.0f;
    h[i] = (BF16)v;
}

__global__ void k_pad_bias(const float* __restrict__ src, float* __restrict__ dst, int n, int np)
{
    int i = threadIdx.x;
    if (i < np) dst[i] = (i < n) ? src[i] : 0.0f;
}

// ---------------- generic batched bf16 WMMA GEMM (per-wave 16x16 tile) ----------------
// flags bit0 = relu, bit1 = write f32 (else bf16).
__global__ void __launch_bounds__(256)
k_gemm_bf16(const BF16* __restrict__ A, const BF16* __restrict__ Bm,
            void* __restrict__ Cp, const float* __restrict__ bias,
            int M, int N, int K, int lda, int ldb, int ldc,
            long long sA, long long sB, long long sC, int batch, int flags)
{
    long wave = ((long)blockIdx.x * blockDim.x + threadIdx.x) >> 5;
    int lane = threadIdx.x & 31;
    int tN = N >> 4;
    long tilesPer = (long)(M >> 4) * tN;
    long total = tilesPer * batch;
    if (wave >= total) return;                 // uniform per wave -> EXEC all-1s inside
    int  b   = (int)(wave / tilesPer);
    long rem = wave - (long)b * tilesPer;
    int  tm  = (int)(rem / tN);
    int  tn  = (int)(rem % tN);
    int  hi  = lane >> 4;                      // 0: K {0..7,16..23}; 1: K {8..15,24..31}
    int  l16 = lane & 15;

    const BF16* arow  = A  + (long)b * sA + (long)(tm * 16 + l16) * lda;
    const BF16* bbase = Bm + (long)b * sB + tn * 16;

    f32x8 acc = {};
    for (int k0 = 0; k0 < K; k0 += 32) {
        ABFrag a;
        a.h8[0] = *(const bf16x8*)(arow + k0 + hi * 8);
        a.h8[1] = *(const bf16x8*)(arow + k0 + 16 + hi * 8);
        bf16x16 bfrag;
        for (int s = 0; s < 16; ++s)
            bfrag[s] = bbase[(long)(k0 + hi * 8 + (s & 7) + (s >> 3) * 16) * ldb + l16];
        acc = __builtin_amdgcn_wmma_f32_16x16x32_bf16(false, a.v, false, bfrag,
                                                      (short)0, acc, false, false);
    }

    int col = tn * 16 + l16;
    float bv = bias ? bias[col] : 0.0f;
    long cb = (long)b * sC;
    for (int r = 0; r < 8; ++r) {
        int row = tm * 16 + r + hi * 8;
        float val = acc[r] + bv;
        if (flags & 1) val = fmaxf(val, 0.0f);
        long idx = cb + (long)row * ldc + col;
        if (flags & 2) ((float*)Cp)[idx] = val;
        else           ((BF16*)Cp)[idx]  = (BF16)val;
    }
}

// ---------------- TDM-staged WMMA GEMM ----------------
// Workgroup macro-tile 128(M) x 64(N). 8 waves, each wave: 16 rows x 64 cols
// (4 accumulators reusing one A fragment). B panel (K x 64, bf16, row-major,
// row stride 64 elem = 128B) staged into LDS once by the Tensor Data Mover.
// Requires M%128==0, N%64==0, K%32==0. flags bit0 = relu, bit1 = f32 out.
__global__ void __launch_bounds__(256)
k_gemm_tdm(const BF16* __restrict__ A, const BF16* __restrict__ Bm,
           void* __restrict__ Cp, const float* __restrict__ bias,
           int M, int N, int K, int lda, int ldb, int ldc, int flags)
{
    extern __shared__ BF16 bpanel[];          // K * 64 bf16
    int nb = N >> 6;
    int bm = (int)blockIdx.x / nb;
    int bn = (int)blockIdx.x % nb;
    int tid = threadIdx.x;
    int w   = tid >> 5;
    int lane = tid & 31;
    int hi  = lane >> 4, l16 = lane & 15;

    if (w == 0) {
        // Build Tensor DMA descriptor (D#) per CDNA5 ISA 8.3/8.4:
        // 2-D tensor, data_size=2B, tile = 64 (x) by K (y), stride0 = ldb.
        unsigned ldsOff = (unsigned)(unsigned long long)(uintptr_t)bpanel;
        unsigned long long ga =
            (unsigned long long)(uintptr_t)(const void*)(Bm + (size_t)bn * 64);
        u32x4 g0;
        g0[0] = 1u;                                    // count=1, user mode
        g0[1] = ldsOff;                                // lds_addr
        g0[2] = (unsigned)(ga & 0xffffffffu);          // global_addr[31:0]
        g0[3] = (unsigned)((ga >> 32) & 0x1ffffffu) | (2u << 30); // addr[56:32] | type=2
        i32x8 g1;
        g1[0] = 0x10000;                               // wg_mask=0, data_size=1 (2B)
        g1[1] = (int)(64u << 16);                      // tensor_dim0[15:0]=64
        g1[2] = (int)((unsigned)(K & 0xffff) << 16);   // dim0 hi=0, tensor_dim1[15:0]=K
        g1[3] = (int)(64u << 16);                      // dim1 hi=0, tile_dim0=64
        g1[4] = (int)(K & 0xffff);                     // tile_dim1=K, tile_dim2=0
        g1[5] = (int)(unsigned)ldb;                    // tensor_dim0_stride[31:0]
        g1[6] = 0;                                     // stride0 hi, stride1 lo
        g1[7] = 0;                                     // stride1 hi
        i32x4 gz  = {0, 0, 0, 0};                      // groups 2/3 unused (2-D)
        i32x8 gz8 = {0, 0, 0, 0, 0, 0, 0, 0};          // extra descriptor words (unused)
        __builtin_amdgcn_tensor_load_to_lds(g0, g1, gz, gz, gz8, 0);
        __builtin_amdgcn_s_wait_tensorcnt(0);
    }
    __syncthreads();                                   // publish panel to all waves

    const BF16* arow = A + ((size_t)bm * 128 + w * 16 + l16) * lda;
    f32x8 acc0 = {}, acc1 = {}, acc2 = {}, acc3 = {};
    for (int k0 = 0; k0 < K; k0 += 32) {
        ABFrag a;
        a.h8[0] = *(const bf16x8*)(arow + k0 + hi * 8);
        a.h8[1] = *(const bf16x8*)(arow + k0 + 16 + hi * 8);
        bf16x16 b0, b1, b2, b3;
        for (int s = 0; s < 16; ++s) {
            const BF16* lrow = bpanel + (size_t)(k0 + hi * 8 + (s & 7) + (s >> 3) * 16) * 64 + l16;
            b0[s] = lrow[0]; b1[s] = lrow[16]; b2[s] = lrow[32]; b3[s] = lrow[48];
        }
        acc0 = __builtin_amdgcn_wmma_f32_16x16x32_bf16(false, a.v, false, b0, (short)0, acc0, false, false);
        acc1 = __builtin_amdgcn_wmma_f32_16x16x32_bf16(false, a.v, false, b1, (short)0, acc1, false, false);
        acc2 = __builtin_amdgcn_wmma_f32_16x16x32_bf16(false, a.v, false, b2, (short)0, acc2, false, false);
        acc3 = __builtin_amdgcn_wmma_f32_16x16x32_bf16(false, a.v, false, b3, (short)0, acc3, false, false);
    }

    f32x8 accs[4] = { acc0, acc1, acc2, acc3 };
    long rowBase = (long)bm * 128 + w * 16 + hi * 8;
#pragma unroll
    for (int j = 0; j < 4; ++j) {
        int col = bn * 64 + j * 16 + l16;
        float bv = bias ? bias[col] : 0.0f;
#pragma unroll
        for (int r = 0; r < 8; ++r) {
            float val = accs[j][r] + bv;
            if (flags & 1) val = fmaxf(val, 0.0f);
            long idx = (rowBase + r) * ldc + col;
            if (flags & 2) ((float*)Cp)[idx] = val;
            else           ((BF16*)Cp)[idx]  = (BF16)val;
        }
    }
}

// ---------------- wave-per-row GEMV (logits = S @ v) ----------------
__global__ void k_gemv(const BF16* __restrict__ A, const float* __restrict__ v,
                       float* __restrict__ out, int rows, int cols, int lda)
{
    long wave = ((long)blockIdx.x * blockDim.x + threadIdx.x) >> 5;
    int lane = threadIdx.x & 31;
    if (wave >= rows) return;
    const BF16* a = A + (long)wave * lda;
    float s = 0.0f;
    for (int c = lane; c < cols; c += 32) s += (float)a[c] * v[c];
    for (int o = 16; o > 0; o >>= 1) s += __shfl_xor(s, o, 32);
    if (lane == 0) out[wave] = s;
}

// ---------------- attention w/ master node (63 real + 1 master) ----------------
__global__ void k_att_master(const float* __restrict__ logits, const BF16* __restrict__ h,
                             float* __restrict__ z, int layerOff)
{
    int g = blockIdx.x;      // 2048 graphs
    int tid = threadIdx.x;   // 256 = H
    __shared__ float sl[64];
    __shared__ float sinv;
    if (tid < 63) sl[tid] = logits[g * 64 + tid];
    __syncthreads();
    if (tid == 0) {
        float m = sl[0];
        for (int i = 1; i < 63; ++i) m = fmaxf(m, sl[i]);
        float s = 0.0f;
        for (int i = 0; i < 63; ++i) { sl[i] = __expf(sl[i] - m); s += sl[i]; }
        sinv = 1.0f / s;
    }
    __syncthreads();
    const BF16* hg = h + (long)g * 64 * 256;
    float accv = 0.0f;
    for (int n = 0; n < 63; ++n) accv += sl[n] * (float)hg[n * 256 + tid];
    z[(long)g * 1024 + layerOff + tid]       = accv * sinv;               // pooled
    z[(long)g * 1024 + layerOff + 256 + tid] = (float)hg[63 * 256 + tid]; // master
}

// ---------------- batch-norm over 2048 rows x 1024 features ----------------
__global__ void k_bn_stats(const float* __restrict__ z, float* __restrict__ mu,
                           float* __restrict__ rsig, int rows, int C)
{
    int c = blockIdx.x, tid = threadIdx.x;  // 256 threads
    float s = 0.0f, s2 = 0.0f;
    for (int r = tid; r < rows; r += 256) {
        float v = z[(long)r * C + c];
        s += v; s2 += v * v;
    }
    __shared__ float sh[256], sh2[256];
    sh[tid] = s; sh2[tid] = s2; __syncthreads();
    for (int o = 128; o > 0; o >>= 1) {
        if (tid < o) { sh[tid] += sh[tid + o]; sh2[tid] += sh2[tid + o]; }
        __syncthreads();
    }
    if (tid == 0) {
        float m = sh[0] / rows;
        float var = sh2[0] / rows - m * m;
        mu[c] = m;
        rsig[c] = rsqrtf(var + 1e-5f);
    }
}

__global__ void k_bn_apply(const float* __restrict__ z, const float* __restrict__ mu,
                           const float* __restrict__ rsig, const float* __restrict__ gamma,
                           const float* __restrict__ beta, BF16* __restrict__ zb,
                           long total, int C)
{
    long i = (long)blockIdx.x * blockDim.x + threadIdx.x;
    if (i >= total) return;
    int c = (int)(i % C);
    zb[i] = (BF16)((z[i] - mu[c]) * rsig[c] * gamma[c] + beta[c]);
}

// ---------------- segment attention pooling (S=16, no master) ----------------
__global__ void k_satt_pool(const float* __restrict__ logits, const BF16* __restrict__ z1,
                            BF16* __restrict__ zp)
{
    int b = blockIdx.x;      // 128
    int tid = threadIdx.x;   // 256
    __shared__ float sl[16];
    __shared__ float sinv;
    if (tid < 16) sl[tid] = logits[b * 16 + tid];
    __syncthreads();
    if (tid == 0) {
        float m = sl[0];
        for (int i = 1; i < 16; ++i) m = fmaxf(m, sl[i]);
        float s = 0.0f;
        for (int i = 0; i < 16; ++i) { sl[i] = __expf(sl[i] - m); s += sl[i]; }
        sinv = 1.0f / s;
    }
    __syncthreads();
    const BF16* zg = z1 + (long)b * 16 * 256;
    float accv = 0.0f;
    for (int n = 0; n < 16; ++n) accv += sl[n] * (float)zg[n * 256 + tid];
    zp[b * 256 + tid] = (BF16)(accv * sinv);
}

// ---------------- final log-softmax (128 rows x 10 classes, ld=16) ----------------
__global__ void k_logsoftmax(const float* __restrict__ logits, float* __restrict__ out)
{
    int r = threadIdx.x;     // 128 rows
    if (r >= 128) return;
    const float* l = logits + r * 16;
    float m = l[0];
    for (int i = 1; i < 10; ++i) m = fmaxf(m, l[i]);
    float s = 0.0f;
    for (int i = 0; i < 10; ++i) s += __expf(l[i] - m);
    float lse = m + __logf(s);
    for (int i = 0; i < 10; ++i) out[r * 10 + i] = l[i] - lse;
}

// ---------------- host launcher ----------------
static inline void* wsp(void* ws, size_t off) { return (void*)((char*)ws + off); }

extern "C" void kernel_launch(void* const* d_in, const int* in_sizes, int n_in,
                              void* d_out, int out_size, void* d_ws, size_t ws_size,
                              hipStream_t stream)
{
    (void)in_sizes; (void)n_in; (void)out_size; (void)ws_size;
    const int  Ff = 300;
    const long Gg = 2048;       // B*S
    const long Tt = Gg * 64;    // 131072 tokens
    const int  Fp = 320;        // F padded to multiple of 32

    const int*   x      = (const int*)  d_in[0];
    const float* adj    = (const float*)d_in[1];
    // d_in[2]=adj_s (unused by reference), d_in[3]=shape0, d_in[4]=shape1
    const float* emb    = (const float*)d_in[5];
    const float* mp0_w1 = (const float*)d_in[6];
    const float* mp0_b1 = (const float*)d_in[7];
    const float* mp0_w2 = (const float*)d_in[8];
    const float* mp0_b2 = (const float*)d_in[9];
    const float* mp1_w1 = (const float*)d_in[10];
    const float* mp1_b1 = (const float*)d_in[11];
    const float* mp1_w2 = (const float*)d_in[12];
    const float* mp1_b2 = (const float*)d_in[13];
    const float* att0_w = (const float*)d_in[14];
    const float* att0_b = (const float*)d_in[15];
    const float* att0_v = (const float*)d_in[16];
    const float* att1_w = (const float*)d_in[17];
    const float* att1_b = (const float*)d_in[18];
    const float* att1_v = (const float*)d_in[19];
    const float* bn_g   = (const float*)d_in[20];
    const float* bn_bt  = (const float*)d_in[21];
    const float* fc1_w  = (const float*)d_in[22];
    const float* fc1_b  = (const float*)d_in[23];
    const float* satt_w = (const float*)d_in[24];
    const float* satt_b = (const float*)d_in[25];
    const float* satt_v = (const float*)d_in[26];
    const float* fc2_w  = (const float*)d_in[27];
    const float* fc2_b  = (const float*)d_in[28];
    const float* fc3_w  = (const float*)d_in[29];
    const float* fc3_b  = (const float*)d_in[30];

    // -------- workspace carve (256B aligned) --------
    size_t off = 0;
    auto take = [&](size_t bytes) { size_t o = off; off += (bytes + 255) & ~(size_t)255; return o; };
    BF16*  hbuf = (BF16*) wsp(d_ws, take((size_t)Tt * Fp * 2));   // h / h'
    BF16*  mbuf = (BF16*) wsp(d_ws, take((size_t)Tt * Fp * 2));   // adj @ h
    BF16*  cbuf = (BF16*) wsp(d_ws, take((size_t)Tt * Fp * 2));   // x1 / att scores
    BF16*  adjb = (BF16*) wsp(d_ws, take((size_t)Gg * 64 * 64 * 2));
    float* zf   = (float*)wsp(d_ws, take((size_t)Gg * 1024 * 4)); // concat reps
    BF16*  zb   = (BF16*) wsp(d_ws, take((size_t)Gg * 1024 * 2)); // normalized z
    BF16*  z1b  = (BF16*) wsp(d_ws, take((size_t)Gg * 256 * 2));  // fc1 out
    BF16*  s2b  = (BF16*) wsp(d_ws, take((size_t)Gg * 256 * 2));  // satt scores
    BF16*  zpb  = (BF16*) wsp(d_ws, take((size_t)128 * 256 * 2)); // satt pooled
    BF16*  y2b  = (BF16*) wsp(d_ws, take((size_t)128 * 128 * 2)); // fc2 out
    float* y3f  = (float*)wsp(d_ws, take((size_t)128 * 16 * 4));  // fc3 logits (f32)
    float* lgt  = (float*)wsp(d_ws, take((size_t)Tt * 4));
    float* lg2  = (float*)wsp(d_ws, take((size_t)Gg * 4));
    float* muv  = (float*)wsp(d_ws, take(1024 * 4));
    float* rsv  = (float*)wsp(d_ws, take(1024 * 4));
    BF16*  w1a  = (BF16*) wsp(d_ws, take(320 * 256 * 2));
    BF16*  w2a  = (BF16*) wsp(d_ws, take(256 * 256 * 2));
    BF16*  w1b  = (BF16*) wsp(d_ws, take(256 * 256 * 2));
    BF16*  w2b  = (BF16*) wsp(d_ws, take(256 * 256 * 2));
    BF16*  awa  = (BF16*) wsp(d_ws, take(256 * 256 * 2));
    BF16*  awb  = (BF16*) wsp(d_ws, take(256 * 256 * 2));
    BF16*  swb  = (BF16*) wsp(d_ws, take(256 * 256 * 2));
    BF16*  f1w  = (BF16*) wsp(d_ws, take(1024 * 256 * 2));
    BF16*  f2w  = (BF16*) wsp(d_ws, take(256 * 128 * 2));
    BF16*  f3w  = (BF16*) wsp(d_ws, take(128 * 16 * 2));
    float* b16  = (float*)wsp(d_ws, take(16 * 4));

    auto cvt = [&](const float* s, BF16* d, int sr, int sc, int dr, int dc) {
        long tot = (long)dr * dc;
        k_convert_pad<<<dim3((unsigned)((tot + 255) / 256)), dim3(256), 0, stream>>>(s, d, sr, sc, dr, dc);
    };
    auto gemm = [&](const BF16* A, const BF16* Bm, void* Cp, const float* bias,
                    int M, int Nc, int K, int lda, int ldb, int ldc,
                    long long sA, long long sB, long long sC, int batch, int flags) {
        long waves = (long)(M >> 4) * (Nc >> 4) * batch;
        k_gemm_bf16<<<dim3((unsigned)((waves + 7) / 8)), dim3(256), 0, stream>>>(
            A, Bm, Cp, bias, M, Nc, K, lda, ldb, ldc, sA, sB, sC, batch, flags);
    };
    auto gemmT = [&](const BF16* A, const BF16* Bm, void* Cp, const float* bias,
                     int M, int Nc, int K, int lda, int ldb, int ldc, int flags) {
        unsigned blocks = (unsigned)((M / 128) * (Nc / 64));
        size_t smem = (size_t)K * 64 * sizeof(BF16);   // B panel in LDS
        k_gemm_tdm<<<dim3(blocks), dim3(256), smem, stream>>>(
            A, Bm, Cp, bias, M, Nc, K, lda, ldb, ldc, flags);
    };
    auto gemv = [&](const BF16* A, const float* v, float* o, long rows, int cols, int lda) {
        k_gemv<<<dim3((unsigned)((rows * 32 + 255) / 256)), dim3(256), 0, stream>>>(
            A, v, o, (int)rows, cols, lda);
    };

    // -------- stage 0: casts + gather --------
    cvt(adj, adjb, (int)(Gg * 64), 64, (int)(Gg * 64), 64);
    cvt(mp0_w1, w1a, Ff, 256, 320, 256);     // pad K 300 -> 320 with zeros
    cvt(mp0_w2, w2a, 256, 256, 256, 256);
    cvt(mp1_w1, w1b, 256, 256, 256, 256);
    cvt(mp1_w2, w2b, 256, 256, 256, 256);
    cvt(att0_w, awa, 256, 256, 256, 256);
    cvt(att1_w, awb, 256, 256, 256, 256);
    cvt(satt_w, swb, 256, 256, 256, 256);
    cvt(fc1_w,  f1w, 1024, 256, 1024, 256);
    cvt(fc2_w,  f2w, 256, 128, 256, 128);
    cvt(fc3_w,  f3w, 128, 10, 128, 16);      // pad N 10 -> 16
    k_pad_bias<<<dim3(1), dim3(16), 0, stream>>>(fc3_b, b16, 10, 16);
    {
        long tot = Tt * (long)Fp;
        k_gather_cast<<<dim3((unsigned)((tot + 255) / 256)), dim3(256), 0, stream>>>(
            x, emb, hbuf, Tt, Ff, Fp);
    }

    // -------- MP layer 0 --------
    // m = adj @ h : batched 64x320x64 (per-wave-tile kernel; M=64 < 128)
    gemm(adjb, hbuf, mbuf, nullptr, 64, 320, 64, 64, 320, 320,
         64 * 64, 64 * 320, 64 * 320, (int)Gg, 0);
    // x1 = relu(m @ w1 + b1) : 131072 x 256, K=320   (TDM-staged weights)
    gemmT(mbuf, w1a, cbuf, mp0_b1, (int)Tt, 256, 320, 320, 256, 256, 1);
    // h' = relu(x1 @ w2 + b2)
    gemmT(cbuf, w2a, hbuf, mp0_b2, (int)Tt, 256, 256, 256, 256, 256, 1);
    // att scores s = relu(h' @ aw + ab)
    gemmT(hbuf, awa, cbuf, att0_b, (int)Tt, 256, 256, 256, 256, 256, 1);
    gemv(cbuf, att0_v, lgt, Tt, 256, 256);
    k_att_master<<<dim3((unsigned)Gg), dim3(256), 0, stream>>>(lgt, hbuf, zf, 0);

    // -------- MP layer 1 (h = h' in hbuf, ld 256) --------
    gemm(adjb, hbuf, mbuf, nullptr, 64, 256, 64, 64, 256, 256,
         64 * 64, 64 * 256, 64 * 256, (int)Gg, 0);
    gemmT(mbuf, w1b, cbuf, mp1_b1, (int)Tt, 256, 256, 256, 256, 256, 1);
    gemmT(cbuf, w2b, hbuf, mp1_b2, (int)Tt, 256, 256, 256, 256, 256, 1);
    gemmT(hbuf, awb, cbuf, att1_b, (int)Tt, 256, 256, 256, 256, 256, 1);
    gemv(cbuf, att1_v, lgt, Tt, 256, 256);
    k_att_master<<<dim3((unsigned)Gg), dim3(256), 0, stream>>>(lgt, hbuf, zf, 512);

    // -------- batch norm + fc1 --------
    k_bn_stats<<<dim3(1024), dim3(256), 0, stream>>>(zf, muv, rsv, (int)Gg, 1024);
    {
        long tot = Gg * 1024;
        k_bn_apply<<<dim3((unsigned)((tot + 255) / 256)), dim3(256), 0, stream>>>(
            zf, muv, rsv, bn_g, bn_bt, zb, tot, 1024);
    }
    gemmT(zb, f1w, z1b, fc1_b, (int)Gg, 256, 1024, 1024, 256, 256, 1);

    // -------- segment attention --------
    gemmT(z1b, swb, s2b, satt_b, (int)Gg, 256, 256, 256, 256, 256, 1);
    gemv(s2b, satt_v, lg2, Gg, 256, 256);
    k_satt_pool<<<dim3(128), dim3(256), 0, stream>>>(lg2, z1b, zpb);

    // -------- head: fc2 -> fc3 -> log_softmax --------
    gemmT(zpb, f2w, y2b, fc2_b, 128, 128, 256, 256, 128, 128, 1);
    gemm(y2b, f3w, y3f, b16, 128, 16, 128, 128, 16, 16, 0, 0, 0, 1, 2);  // f32 out
    k_logsoftmax<<<dim3(1), dim3(128), 0, stream>>>(y3f, (float*)d_out);
}
